// CausalAttention_59803124629884
// MI455X (gfx1250) — matrix operand order repaired and checked by reading, not compiled, not run on hardware
//
#include <hip/hip_runtime.h>
#include <hip/hip_bf16.h>

#define TB 2
#define TT 4096
#define TC 768
#define TH 12
#define TD 64
#define TM (TB * TT)   // 8192 rows

typedef __attribute__((ext_vector_type(16))) __bf16 bf16x16;
typedef __attribute__((ext_vector_type(8)))  __bf16 bf16x8;
typedef __attribute__((ext_vector_type(8)))  float  f32x8;

__device__ __forceinline__ f32x8 wmma_bf16f32(bf16x16 a, bf16x16 b, f32x8 c) {
  // (neg_a, A, neg_b, B, c_mod, C, reuse_a, reuse_b)
  return __builtin_amdgcn_wmma_f32_16x16x32_bf16(false, a, false, b, (short)0, c, false, false);
}

__device__ __forceinline__ f32x8 zero8() {
  f32x8 z = {0.f, 0.f, 0.f, 0.f, 0.f, 0.f, 0.f, 0.f};
  return z;
}

// A-fragment 16x32 bf16 from row-major A[row][k], leading dim ld.
// Lane l: row = l&15; lanes<16 hold K {0..7,16..23}+k0, lanes>=16 hold K {8..15,24..31}+k0.
__device__ __forceinline__ bf16x16 load_a_frag(const __bf16* A, int ld, int row0, int k0) {
  const int l  = threadIdx.x & 31;
  const int m  = l & 15;
  const int kb = (l < 16) ? 0 : 8;
  const __bf16* p = A + (size_t)(row0 + m) * ld + (k0 + kb);
  bf16x8 x0 = *(const bf16x8*)(p);        // K = k0+kb+0..7
  bf16x8 x1 = *(const bf16x8*)(p + 16);   // K = k0+16+kb+0..7
  bf16x16 r;
#pragma unroll
  for (int i = 0; i < 8; ++i) { r[i] = x0[i]; r[i + 8] = x1[i]; }
  return r;
}

// B-fragment 32x16 bf16 where memory holds B^T row-major: Bt[n][k], leading dim ld.
// Lane l: col = l&15; lanes<16 hold K {0..7,16..23}+k0, lanes>=16 hold K {8..15,24..31}+k0.
__device__ __forceinline__ bf16x16 load_bT_frag(const __bf16* Bt, int ld, int col0, int k0) {
  const int l  = threadIdx.x & 31;
  const int c  = l & 15;
  const int kk = k0 + ((l < 16) ? 0 : 8);
  const __bf16* p = Bt + (size_t)(col0 + c) * ld + kk;
  bf16x8 x0 = *(const bf16x8*)(p);
  bf16x8 x1 = *(const bf16x8*)(p + 16);
  bf16x16 r;
#pragma unroll
  for (int i = 0; i < 8; ++i) { r[i] = x0[i]; r[i + 8] = x1[i]; }
  return r;
}

// ---------------- prep: fp32 -> bf16 (x), fp32 W -> bf16 W^T ----------------

__global__ void prep_x_kernel(const float* __restrict__ x, __bf16* __restrict__ xbf) {
  const int i = blockIdx.x * 256 + threadIdx.x;
  xbf[i] = (__bf16)x[i];
}

__global__ void prep_w_kernel(const float* __restrict__ Wq, const float* __restrict__ Wk,
                              const float* __restrict__ Wv, const float* __restrict__ Wo,
                              __bf16* __restrict__ WtQ, __bf16* __restrict__ WtK,
                              __bf16* __restrict__ WtV, __bf16* __restrict__ WtO) {
  const int i = blockIdx.x * 256 + threadIdx.x;  // over TC*TC
  const float* W;
  __bf16* Wt;
  switch (blockIdx.y) {
    case 0:  W = Wq; Wt = WtQ; break;
    case 1:  W = Wk; Wt = WtK; break;
    case 2:  W = Wv; Wt = WtV; break;
    default: W = Wo; Wt = WtO; break;
  }
  const int k = i / TC;      // input-dim row of W
  const int n = i % TC;      // output column
  Wt[(size_t)n * TC + k] = (__bf16)W[i];
}

// ---------------- QKV projection GEMM (bf16 WMMA, f32 accum) ----------------
// grid(TM/256, TH, 3) block 256. Wave tile 32x64 (one head wide): 2 A-frags share
// each B-frag -> 8 wmma per 12 b128 loads per 32-deep k-step.

__global__ void __launch_bounds__(256)
qkv_gemm_kernel(const __bf16* __restrict__ xbf,
                const __bf16* __restrict__ WtQ, const __bf16* __restrict__ WtK,
                const __bf16* __restrict__ WtV,
                const float* __restrict__ bq, const float* __restrict__ bk,
                const float* __restrict__ bv,
                __bf16* __restrict__ Qo, __bf16* __restrict__ Ko, __bf16* __restrict__ Vto) {
  const int wave = threadIdx.x >> 5;
  const int lane = threadIdx.x & 31;
  const int rowBase = blockIdx.x * 256 + wave * 32;
  const int h = blockIdx.y;
  const int colBase = h * 64;
  const int z = blockIdx.z;
  const __bf16* Wt  = (z == 0) ? WtQ : (z == 1) ? WtK : WtV;
  const float* bias = (z == 0) ? bq  : (z == 1) ? bk  : bv;

  f32x8 acc[2][4];
#pragma unroll
  for (int i = 0; i < 2; ++i)
#pragma unroll
    for (int j = 0; j < 4; ++j) acc[i][j] = zero8();

  for (int k0 = 0; k0 < TC; k0 += 32) {
    bf16x16 a0 = load_a_frag(xbf, TC, rowBase, k0);
    bf16x16 a1 = load_a_frag(xbf, TC, rowBase + 16, k0);
#pragma unroll
    for (int j = 0; j < 4; ++j) {
      bf16x16 bfr = load_bT_frag(Wt, TC, colBase + j * 16, k0);
      acc[0][j] = wmma_bf16f32(a0, bfr, acc[0][j]);
      acc[1][j] = wmma_bf16f32(a1, bfr, acc[1][j]);
    }
  }

  const int halfSel = lane >> 4;
  const int colL = lane & 15;
#pragma unroll
  for (int i = 0; i < 2; ++i) {
#pragma unroll
    for (int j = 0; j < 4; ++j) {
      const int d = j * 16 + colL;
      const float bb = bias[colBase + d];
#pragma unroll
      for (int r = 0; r < 8; ++r) {
        const int g = rowBase + i * 16 + r + 8 * halfSel;
        const int bi = g / TT;
        const int t  = g % TT;
        const float val = acc[i][j][r] + bb;
        if (z == 2) {
          // V stored transposed: [B,H,64,T]
          Vto[(((size_t)bi * TH + h) * TD + d) * TT + t] = (__bf16)val;
        } else {
          __bf16* O = (z == 0) ? Qo : Ko;
          O[(((size_t)bi * TH + h) * TT + t) * TD + d] = (__bf16)val;
        }
      }
    }
  }
}

// ---------------- causal flash attention ----------------
// grid(TT/256, TH, TB) block 256. Each wave: 32 query rows, online softmax.
// K^T B-frags reused across both 16-row query sub-tiles; V B-frags feed 2 wmma each.

__global__ void __launch_bounds__(256)
flash_attn_kernel(const __bf16* __restrict__ Q, const __bf16* __restrict__ K,
                  const __bf16* __restrict__ Vt, __bf16* __restrict__ AO) {
  __shared__ __bf16 Plds[8][32 * 32];  // per-wave P staging (C-layout -> A-frag layout)
  const int wave = threadIdx.x >> 5;
  const int lane = threadIdx.x & 31;
  const int halfSel = lane >> 4;
  const int colL = lane & 15;
  const int b = blockIdx.z;
  const int h = blockIdx.y;
  const int qBase = blockIdx.x * 256 + wave * 32;

  const __bf16* Qh  = Q  + ((size_t)(b * TH + h)) * TT * TD;
  const __bf16* Kh  = K  + ((size_t)(b * TH + h)) * TT * TD;
  const __bf16* Vth = Vt + ((size_t)(b * TH + h)) * TD * TT;  // [64][T]

  bf16x16 qa[2][2];
#pragma unroll
  for (int i = 0; i < 2; ++i) {
    qa[i][0] = load_a_frag(Qh, TD, qBase + i * 16, 0);
    qa[i][1] = load_a_frag(Qh, TD, qBase + i * 16, 32);
  }

  f32x8 o[2][4];
#pragma unroll
  for (int i = 0; i < 2; ++i)
#pragma unroll
    for (int j = 0; j < 4; ++j) o[i][j] = zero8();
  float rm[2][8], rl[2][8];
#pragma unroll
  for (int i = 0; i < 2; ++i)
#pragma unroll
    for (int r = 0; r < 8; ++r) { rm[i][r] = -3.0e38f; rl[i][r] = 0.f; }

  const float sc = 0.125f * 1.44269504088896340736f;  // (1/sqrt(64)) * log2(e)
  __bf16* myP = &Plds[wave][0];

  for (int kb = 0; kb <= qBase + 31; kb += 32) {
    f32x8 s[2][2];
#pragma unroll
    for (int i = 0; i < 2; ++i) { s[i][0] = zero8(); s[i][1] = zero8(); }
    {
      bf16x16 b00 = load_bT_frag(Kh, TD, kb, 0);
      bf16x16 b01 = load_bT_frag(Kh, TD, kb, 32);
      bf16x16 b10 = load_bT_frag(Kh, TD, kb + 16, 0);
      bf16x16 b11 = load_bT_frag(Kh, TD, kb + 16, 32);
#pragma unroll
      for (int i = 0; i < 2; ++i) {
        s[i][0] = wmma_bf16f32(qa[i][0], b00, s[i][0]);
        s[i][0] = wmma_bf16f32(qa[i][1], b01, s[i][0]);
        s[i][1] = wmma_bf16f32(qa[i][0], b10, s[i][1]);
        s[i][1] = wmma_bf16f32(qa[i][1], b11, s[i][1]);
      }
    }
#pragma unroll
    for (int i = 0; i < 2; ++i) {
#pragma unroll
      for (int r = 0; r < 8; ++r) {
        const int row = qBase + i * 16 + r + 8 * halfSel;
        const int c0 = kb + colL;
        const int c1 = c0 + 16;
        float v0 = (c0 <= row) ? s[i][0][r] * sc : -3.0e38f;
        float v1 = (c1 <= row) ? s[i][1][r] * sc : -3.0e38f;
        // row max over 32 keys: reduce across the 16-lane half (xor masks < 16)
        float bm = fmaxf(v0, v1);
#pragma unroll
        for (int off = 1; off < 16; off <<= 1) bm = fmaxf(bm, __shfl_xor(bm, off, 32));
        const float mnew  = fmaxf(rm[i][r], bm);
        const float alpha = exp2f(rm[i][r] - mnew);
        const float p0 = exp2f(v0 - mnew);
        const float p1 = exp2f(v1 - mnew);
        float ps = p0 + p1;
#pragma unroll
        for (int off = 1; off < 16; off <<= 1) ps += __shfl_xor(ps, off, 32);
        rl[i][r] = rl[i][r] * alpha + ps;
        rm[i][r] = mnew;
#pragma unroll
        for (int j = 0; j < 4; ++j) o[i][j][r] *= alpha;
        const int lr = i * 16 + r + 8 * halfSel;  // local row 0..31
        myP[lr * 32 + colL]      = (__bf16)p0;
        myP[lr * 32 + colL + 16] = (__bf16)p1;
      }
    }
    // same-wave LDS ordering: DS ops are in-order; wait before re-reading as A-frag
    asm volatile("s_wait_dscnt 0" ::: "memory");
    const bf16x16 pa0 = load_a_frag(myP, 32, 0, 0);
    const bf16x16 pa1 = load_a_frag(myP, 32, 16, 0);
#pragma unroll
    for (int j = 0; j < 4; ++j) {
      bf16x16 vb = load_bT_frag(Vth, TT, j * 16, kb);  // B = V[32 keys][16 d] via Vt
      o[0][j] = wmma_bf16f32(pa0, vb, o[0][j]);
      o[1][j] = wmma_bf16f32(pa1, vb, o[1][j]);
    }
  }

#pragma unroll
  for (int i = 0; i < 2; ++i) {
#pragma unroll
    for (int r = 0; r < 8; ++r) {
      const float inv = 1.0f / rl[i][r];
      const int t = qBase + i * 16 + r + 8 * halfSel;
#pragma unroll
      for (int j = 0; j < 4; ++j) {
        AO[((size_t)b * TT + t) * TC + h * TD + j * 16 + colL] = (__bf16)(o[i][j][r] * inv);
      }
    }
  }
}

// ---------------- output projection GEMM -> fp32 d_out ----------------
// grid(TM/256, TC/64) block 256. Wave tile 32x64.

__global__ void __launch_bounds__(256)
out_gemm_kernel(const __bf16* __restrict__ A, const __bf16* __restrict__ WtO,
                const float* __restrict__ bo, float* __restrict__ out) {
  const int wave = threadIdx.x >> 5;
  const int lane = threadIdx.x & 31;
  const int rowBase = blockIdx.x * 256 + wave * 32;
  const int colBase = blockIdx.y * 64;

  f32x8 acc[2][4];
#pragma unroll
  for (int i = 0; i < 2; ++i)
#pragma unroll
    for (int j = 0; j < 4; ++j) acc[i][j] = zero8();

  for (int k0 = 0; k0 < TC; k0 += 32) {
    bf16x16 a0 = load_a_frag(A, TC, rowBase, k0);
    bf16x16 a1 = load_a_frag(A, TC, rowBase + 16, k0);
#pragma unroll
    for (int j = 0; j < 4; ++j) {
      bf16x16 bfr = load_bT_frag(WtO, TC, colBase + j * 16, k0);
      acc[0][j] = wmma_bf16f32(a0, bfr, acc[0][j]);
      acc[1][j] = wmma_bf16f32(a1, bfr, acc[1][j]);
    }
  }

  const int halfSel = lane >> 4;
  const int colL = lane & 15;
#pragma unroll
  for (int i = 0; i < 2; ++i) {
#pragma unroll
    for (int j = 0; j < 4; ++j) {
      const int col = colBase + j * 16 + colL;
      const float bb = bo[col];
#pragma unroll
      for (int r = 0; r < 8; ++r) {
        const int g = rowBase + i * 16 + r + 8 * halfSel;
        out[(size_t)g * TC + col] = acc[i][j][r] + bb;
      }
    }
  }
}

// ---------------- host launcher ----------------

extern "C" void kernel_launch(void* const* d_in, const int* in_sizes, int n_in,
                              void* d_out, int out_size, void* d_ws, size_t ws_size,
                              hipStream_t stream) {
  (void)in_sizes; (void)n_in; (void)out_size; (void)ws_size;
  const float* x  = (const float*)d_in[0];
  const float* Wq = (const float*)d_in[1];
  const float* bq = (const float*)d_in[2];
  const float* Wk = (const float*)d_in[3];
  const float* bk = (const float*)d_in[4];
  const float* Wv = (const float*)d_in[5];
  const float* bv = (const float*)d_in[6];
  const float* Wo = (const float*)d_in[7];
  const float* bo = (const float*)d_in[8];
  float* out = (float*)d_out;

  char* ws = (char*)d_ws;
  size_t off = 0;
  auto take = [&](size_t bytes) -> char* {
    off = (off + 255) & ~(size_t)255;
    char* p = ws + off;
    off += bytes;
    return p;
  };
  __bf16* xbf = (__bf16*)take((size_t)TM * TC * 2);
  __bf16* WtQ = (__bf16*)take((size_t)TC * TC * 2);
  __bf16* WtK = (__bf16*)take((size_t)TC * TC * 2);
  __bf16* WtV = (__bf16*)take((size_t)TC * TC * 2);
  __bf16* WtO = (__bf16*)take((size_t)TC * TC * 2);
  __bf16* Qb  = (__bf16*)take((size_t)TB * TH * TT * TD * 2);
  __bf16* Kb  = (__bf16*)take((size_t)TB * TH * TT * TD * 2);
  __bf16* Vtb = (__bf16*)take((size_t)TB * TH * TD * TT * 2);
  __bf16* AO  = (__bf16*)take((size_t)TM * TC * 2);

  prep_x_kernel<<<(TM * TC) / 256, 256, 0, stream>>>(x, xbf);
  prep_w_kernel<<<dim3((TC * TC) / 256, 4), 256, 0, stream>>>(Wq, Wk, Wv, Wo,
                                                              WtQ, WtK, WtV, WtO);
  qkv_gemm_kernel<<<dim3(TM / 256, TH, 3), 256, 0, stream>>>(
      xbf, WtQ, WtK, WtV, bq, bk, bv, Qb, Kb, Vtb);
  flash_attn_kernel<<<dim3(TT / 256, TH, TB), 256, 0, stream>>>(Qb, Kb, Vtb, AO);
  out_gemm_kernel<<<dim3(TM / 256, TC / 64), 256, 0, stream>>>(AO, WtO, bo, out);
}
